// TransformerBlock_38474317037959
// MI455X (gfx1250) — matrix operand order repaired and checked by reading, not compiled, not run on hardware
//
#include <hip/hip_runtime.h>
#include <hip/hip_bf16.h>

typedef __bf16 bf16;
typedef __attribute__((ext_vector_type(16))) __bf16 v16bf;
typedef __attribute__((ext_vector_type(8)))  __bf16 v8bf;
typedef __attribute__((ext_vector_type(8)))  float  v8f;

#define DIM 2048
#define NH 16
#define NKV 4
#define HD 128
#define HID 8192
#define BATCH 2
#define SEQ 2048
#define ROWS (BATCH*SEQ)   // 4096

// Build a 16-element bf16 fragment from two aligned 16-byte chunks.
__device__ __forceinline__ v16bf ldfrag(const bf16* p0, const bf16* p1) {
    v8bf lo = *(const v8bf*)p0;
    v8bf hi = *(const v8bf*)p1;
    return __builtin_shufflevector(lo, hi, 0,1,2,3,4,5,6,7,8,9,10,11,12,13,14,15);
}

// ---------------------------------------------------------------------------
// fp32 [K][N] -> bf16 [N][K] transposing convert (one-time weight prep)
// ---------------------------------------------------------------------------
__global__ void cvt_transpose_bf16(const float* __restrict__ in, bf16* __restrict__ out,
                                   int K, int N) {
    long idx = (long)blockIdx.x * blockDim.x + threadIdx.x;
    long total = (long)K * N;
    long stride = (long)gridDim.x * blockDim.x;
    for (; idx < total; idx += stride) {
        int  n = (int)(idx % N);
        long k = idx / N;
        out[(size_t)n * K + k] = (bf16)in[idx];   // coalesced read, scattered write
    }
}

// ---------------------------------------------------------------------------
// RMSNorm: one block (256 thr) per row of [ROWS, DIM]; bf16 output
// ---------------------------------------------------------------------------
__global__ __launch_bounds__(256) void rmsnorm_bf16(const float* __restrict__ x,
                                                    const float* __restrict__ w,
                                                    bf16* __restrict__ out) {
    int row = blockIdx.x;
    const float* xr = x + (size_t)row * DIM;
    float vals[8];
    float ss = 0.f;
#pragma unroll
    for (int i = 0; i < 8; ++i) {
        int c = threadIdx.x + i * 256;
        float v = xr[c];
        vals[i] = v;
        ss += v * v;
    }
#pragma unroll
    for (int off = 16; off > 0; off >>= 1) ss += __shfl_xor(ss, off, 32);
    __shared__ float red[8];
    int wv = threadIdx.x >> 5, lane = threadIdx.x & 31;
    if (lane == 0) red[wv] = ss;
    __syncthreads();
    __shared__ float stot;
    if (threadIdx.x == 0) {
        float s = 0.f;
        for (int i = 0; i < 8; ++i) s += red[i];
        stot = s;
    }
    __syncthreads();
    float inv = rsqrtf(stot / (float)DIM + 1e-5f);
#pragma unroll
    for (int i = 0; i < 8; ++i) {
        int c = threadIdx.x + i * 256;
        out[(size_t)row * DIM + c] = (bf16)(w[c] * vals[i] * inv);
    }
}

// ---------------------------------------------------------------------------
// RoPE: in [B,S,nh,128] f32  ->  out [B,nh,S,128] bf16 (head-major for attn)
// ---------------------------------------------------------------------------
__global__ void rope_kernel(const float* __restrict__ in,
                            const float* __restrict__ fcos,
                            const float* __restrict__ fsin,
                            bf16* __restrict__ out, int nh) {
    long idx = (long)blockIdx.x * blockDim.x + threadIdx.x;
    long total = (long)BATCH * SEQ * nh * 64;
    if (idx >= total) return;
    int p = idx & 63;
    int h = (idx >> 6) % nh;
    int s = (int)((idx / ((long)64 * nh)) % SEQ);
    int b = (int)(idx / ((long)64 * nh * SEQ));
    const float* base = in + (((size_t)b * SEQ + s) * nh + h) * HD;
    float xr = base[2 * p], xi = base[2 * p + 1];
    float c = fcos[s * 64 + p], sn = fsin[s * 64 + p];
    bf16* ob = out + (((size_t)b * nh + h) * SEQ + s) * HD;
    ob[2 * p]     = (bf16)(xr * c - xi * sn);
    ob[2 * p + 1] = (bf16)(xr * sn + xi * c);
}

// V: [B,S,4,128] f32 -> [B,4,128,S] bf16 (d-major so P@V B-frags are contiguous)
__global__ void vconv_kernel(const float* __restrict__ in, bf16* __restrict__ out) {
    long idx = (long)blockIdx.x * blockDim.x + threadIdx.x;
    long total = (long)BATCH * SEQ * NKV * HD;
    if (idx >= total) return;
    int d = idx & 127;
    int h = (idx >> 7) & 3;
    int s = (int)((idx >> 9) % SEQ);
    int b = (int)(idx / ((long)512 * SEQ));
    out[(((size_t)b * NKV + h) * HD + d) * SEQ + s] =
        (bf16)in[(((size_t)b * SEQ + s) * NKV + h) * HD + d];
}

// ---------------------------------------------------------------------------
// Tiled bf16 WMMA GEMM: C[M,N] = A[M,K] @ Bt[N,K]^T  (Bt = B transposed).
// Block tile 256x128, 8 waves in 4x2, each wave 64x64 (4x4 WMMA tiles).
// Per k-step: 16 ds_load_b128 feed 16 v_wmma (1.0 loads/WMMA).
// Software-pipelined: next k-tile global loads overlap WMMA compute.
// emode: 0 -> outF = acc
//        1 -> outF = acc + aux            (residual)
//        2 -> outH = bf16(acc)
//        3 -> outH = bf16(silu(aux)*acc)  (fused SwiGLU)
// ---------------------------------------------------------------------------
__global__ __launch_bounds__(256) void gemm_bf16(const bf16* __restrict__ A,
                                                 const bf16* __restrict__ Bt,
                                                 int M, int N, int K, int emode,
                                                 const float* __restrict__ aux,
                                                 float* __restrict__ outF,
                                                 bf16*  __restrict__ outH) {
    __shared__ bf16 As[256][40];    // row stride 80B (16B aligned)
    __shared__ bf16 Bs[128][40];    // Bt tile, row = n, col = k
    int tid = threadIdx.x;
    int lane = tid & 31, w = tid >> 5;
    int hi = lane >> 4, lr = lane & 15;
    int m0 = blockIdx.y * 256, n0 = blockIdx.x * 128;
    int wm = (w & 3) * 64, wn = (w >> 2) * 64;

    // cooperative-load chunk coords: 4 consecutive lanes cover one row (32 k)
    int ar[4], ac[4], br[2], bc[2];
#pragma unroll
    for (int ch = 0; ch < 4; ++ch) {           // A tile: 256 rows x 32 k = 1024 chunks
        int e = ch * 256 + tid;
        ar[ch] = e >> 2; ac[ch] = (e & 3) * 8;
    }
#pragma unroll
    for (int ch = 0; ch < 2; ++ch) {           // B tile: 128 rows x 32 k = 512 chunks
        int e = ch * 256 + tid;
        br[ch] = e >> 2; bc[ch] = (e & 3) * 8;
    }

    v8f acc[4][4];
#pragma unroll
    for (int mt = 0; mt < 4; ++mt)
#pragma unroll
        for (int nt = 0; nt < 4; ++nt)
#pragma unroll
            for (int v = 0; v < 8; ++v) acc[mt][nt][v] = 0.f;

    // prefetch first k-tile into registers
    v8bf ra[4], rb[2];
#pragma unroll
    for (int ch = 0; ch < 4; ++ch)
        ra[ch] = *(const v8bf*)(A + (size_t)(m0 + ar[ch]) * K + ac[ch]);
#pragma unroll
    for (int ch = 0; ch < 2; ++ch)
        rb[ch] = *(const v8bf*)(Bt + (size_t)(n0 + br[ch]) * K + bc[ch]);

    for (int k0 = 0; k0 < K; k0 += 32) {
        __syncthreads();                      // prev-iter readers done
#pragma unroll
        for (int ch = 0; ch < 4; ++ch) *(v8bf*)&As[ar[ch]][ac[ch]] = ra[ch];
#pragma unroll
        for (int ch = 0; ch < 2; ++ch) *(v8bf*)&Bs[br[ch]][bc[ch]] = rb[ch];
        __syncthreads();

        if (k0 + 32 < K) {                    // overlap next global loads with WMMA
            int kn = k0 + 32;
#pragma unroll
            for (int ch = 0; ch < 4; ++ch)
                ra[ch] = *(const v8bf*)(A + (size_t)(m0 + ar[ch]) * K + kn + ac[ch]);
#pragma unroll
            for (int ch = 0; ch < 2; ++ch)
                rb[ch] = *(const v8bf*)(Bt + (size_t)(n0 + br[ch]) * K + kn + bc[ch]);
        }

        v16bf af[4], bfr[4];
#pragma unroll
        for (int mt = 0; mt < 4; ++mt) {      // A frag: row=lane&15, chunks at hi*8, 16+hi*8
            const bf16* p = &As[wm + mt * 16 + lr][hi * 8];
            af[mt] = ldfrag(p, p + 16);
        }
#pragma unroll
        for (int nt = 0; nt < 4; ++nt) {      // B frag: col(n)=lane&15, chunks at hi*8, 16+hi*8
            const bf16* p = &Bs[wn + nt * 16 + lr][hi * 8];
            bfr[nt] = ldfrag(p, p + 16);
        }
#pragma unroll
        for (int mt = 0; mt < 4; ++mt)
#pragma unroll
            for (int nt = 0; nt < 4; ++nt)
                acc[mt][nt] = __builtin_amdgcn_wmma_f32_16x16x32_bf16(
                    false, af[mt], false, bfr[nt], (short)0, acc[mt][nt], false, false);
    }

#pragma unroll
    for (int mt = 0; mt < 4; ++mt)
#pragma unroll
        for (int nt = 0; nt < 4; ++nt)
#pragma unroll
            for (int v = 0; v < 8; ++v) {
                int row = m0 + wm + mt * 16 + v + hi * 8;
                int col = n0 + wn + nt * 16 + lr;
                size_t idx = (size_t)row * N + col;
                float a = acc[mt][nt][v];
                if (emode == 0) {
                    outF[idx] = a;
                } else if (emode == 1) {
                    outF[idx] = a + aux[idx];
                } else if (emode == 2) {
                    outH[idx] = (bf16)a;
                } else {
                    float g = aux[idx];
                    float sg = g / (1.f + __expf(-g));   // silu(g@w1)
                    outH[idx] = (bf16)(sg * a);
                }
            }
}

// ---------------------------------------------------------------------------
// Flash attention (causal, GQA rep=4).
// Q [B,NH,S,128], K [B,NKV,S,128], V [B,NKV,128,S]  (all bf16).
// Block = (128 q rows, one head); 8 waves x 16 q rows; key tiles of 32.
// QK^T and P@V via v_wmma_f32_16x16x32_bf16. Output bf16 [B,S,NH*128].
// ---------------------------------------------------------------------------
__global__ __launch_bounds__(256) void attn_kernel(const bf16* __restrict__ Qb,
                                                   const bf16* __restrict__ Kb,
                                                   const bf16* __restrict__ Vb,
                                                   bf16* __restrict__ Ob) {
    int qblk = blockIdx.x, h = blockIdx.y, b = blockIdx.z;
    int kvh = h >> 2;
    int tid = threadIdx.x, lane = tid & 31, w = tid >> 5;
    int hi = lane >> 4, lr = lane & 15;
    int q0 = qblk * 128;

    __shared__ bf16 Kt[32][136];        // [key][d], row stride 272B
    __shared__ bf16 Vt[128][40];        // [d][key], row stride 80B
    __shared__ bf16 Pw[8][16][40];      // per-wave P scratch, row stride 80B

    // preload Q fragments (4 k-steps of 32 over HEAD_DIM=128)
    v16bf aQ[4];
    {
        const bf16* qptr = Qb + (((size_t)b * NH + h) * SEQ + (q0 + w * 16 + lr)) * HD;
#pragma unroll
        for (int s4 = 0; s4 < 4; ++s4) {
            const bf16* p = qptr + s4 * 32 + hi * 8;
            aQ[s4] = ldfrag(p, p + 16);
        }
    }

    v8f O[8];
    float mrun[8], lrun[8];
#pragma unroll
    for (int dt = 0; dt < 8; ++dt)
#pragma unroll
        for (int v = 0; v < 8; ++v) O[dt][v] = 0.f;
#pragma unroll
    for (int v = 0; v < 8; ++v) { mrun[v] = -3.0e38f; lrun[v] = 0.f; }

    const bf16* kbase = Kb + ((size_t)b * NKV + kvh) * SEQ * HD;        // [key][d]
    const bf16* vbase = Vb + ((size_t)b * NKV + kvh) * (size_t)HD * SEQ; // [d][key]
    const float scale = 0.08838834764831845f;   // 1/sqrt(128)
    int nkt = q0 / 32 + 4;                       // causal: keys < q0+128

    // per-thread chunk coords for the cooperative tile loads
    int e0 = tid * 2, e1 = tid * 2 + 1;
    int kr0 = e0 >> 4, kc0 = (e0 & 15) * 8;      // K tile: 32 rows x 128 d
    int kr1 = e1 >> 4, kc1 = (e1 & 15) * 8;
    int vr0 = e0 >> 2, vc0 = (e0 & 3) * 8;       // V tile: 128 d x 32 keys
    int vr1 = e1 >> 2, vc1 = (e1 & 3) * 8;

    for (int kt = 0; kt < nkt; ++kt) {
        __syncthreads();
        *(v8bf*)&Kt[kr0][kc0] = *(const v8bf*)(kbase + (size_t)(kt * 32 + kr0) * HD + kc0);
        *(v8bf*)&Kt[kr1][kc1] = *(const v8bf*)(kbase + (size_t)(kt * 32 + kr1) * HD + kc1);
        *(v8bf*)&Vt[vr0][vc0] = *(const v8bf*)(vbase + (size_t)vr0 * SEQ + kt * 32 + vc0);
        *(v8bf*)&Vt[vr1][vc1] = *(const v8bf*)(vbase + (size_t)vr1 * SEQ + kt * 32 + vc1);
        __syncthreads();

        // scores: 16x32 per wave = 2 n-tiles, K over d in 4 steps of 32
        v8f sc[2];
#pragma unroll
        for (int nt = 0; nt < 2; ++nt) {
#pragma unroll
            for (int v = 0; v < 8; ++v) sc[nt][v] = 0.f;
            int krow = nt * 16 + lr;             // key within tile (B-matrix col)
#pragma unroll
            for (int s4 = 0; s4 < 4; ++s4) {
                const bf16* p = &Kt[krow][s4 * 32 + hi * 8];   // chunks at hi*8, 16+hi*8 in d
                v16bf bK = ldfrag(p, p + 16);
                sc[nt] = __builtin_amdgcn_wmma_f32_16x16x32_bf16(
                    false, aQ[s4], false, bK, (short)0, sc[nt], false, false);
            }
        }

        // online softmax per owned row
#pragma unroll
        for (int v = 0; v < 8; ++v) {
            int row_abs = q0 + w * 16 + v + hi * 8;
            float s0 = sc[0][v] * scale, s1 = sc[1][v] * scale;
            if (kt * 32 + lr > row_abs)      s0 = -1e30f;
            if (kt * 32 + 16 + lr > row_abs) s1 = -1e30f;
            float m = fmaxf(s0, s1);
#pragma unroll
            for (int off = 1; off < 16; off <<= 1) m = fmaxf(m, __shfl_xor(m, off, 32));
            float mnew = fmaxf(mrun[v], m);
            float alpha = __expf(mrun[v] - mnew);
            float p0 = __expf(s0 - mnew), p1 = __expf(s1 - mnew);
            float rs = p0 + p1;
#pragma unroll
            for (int off = 1; off < 16; off <<= 1) rs += __shfl_xor(rs, off, 32);
            lrun[v] = lrun[v] * alpha + rs;
            mrun[v] = mnew;
#pragma unroll
            for (int dt = 0; dt < 8; ++dt) O[dt][v] *= alpha;
            Pw[w][v + hi * 8][lr]      = (bf16)p0;   // C-layout -> row-major P
            Pw[w][v + hi * 8][16 + lr] = (bf16)p1;
        }
        // per-wave LDS store->load ordering before the A-layout gather
        asm volatile("s_wait_dscnt 0" ::: "memory");

        v16bf aP;                                 // P as A-matrix 16x32
        {
            const bf16* p = &Pw[w][lr][hi * 8];
            aP = ldfrag(p, p + 16);
        }
#pragma unroll
        for (int dt = 0; dt < 8; ++dt) {          // O += P @ V
            const bf16* p = &Vt[dt * 16 + lr][hi * 8];   // chunks at hi*8, 16+hi*8 in key
            v16bf bV = ldfrag(p, p + 16);
            O[dt] = __builtin_amdgcn_wmma_f32_16x16x32_bf16(
                false, aP, false, bV, (short)0, O[dt], false, false);
        }
    }

    // write O / l  to [B,S,NH*128] bf16
#pragma unroll
    for (int dt = 0; dt < 8; ++dt)
#pragma unroll
        for (int v = 0; v < 8; ++v) {
            int row = q0 + w * 16 + v + hi * 8;
            float val = O[dt][v] / lrun[v];
            Ob[((size_t)b * SEQ + row) * (NH * HD) + h * HD + dt * 16 + lr] = (bf16)val;
        }
}

// ---------------------------------------------------------------------------
// Host launcher
// ---------------------------------------------------------------------------
static inline void* wsalloc(char*& p, size_t bytes) {
    void* r = p;
    p += (bytes + 255) & ~(size_t)255;
    return r;
}

extern "C" void kernel_launch(void* const* d_in, const int* in_sizes, int n_in,
                              void* d_out, int out_size, void* d_ws, size_t ws_size,
                              hipStream_t stream) {
    const float* x    = (const float*)d_in[0];
    const float* wq   = (const float*)d_in[1];
    const float* wk   = (const float*)d_in[2];
    const float* wv   = (const float*)d_in[3];
    const float* wo   = (const float*)d_in[4];
    const float* w1   = (const float*)d_in[5];
    const float* w2   = (const float*)d_in[6];
    const float* w3   = (const float*)d_in[7];
    const float* anw  = (const float*)d_in[8];
    const float* fnw  = (const float*)d_in[9];
    const float* fcos = (const float*)d_in[10];
    const float* fsin = (const float*)d_in[11];
    // d_in[12] = mask (causality handled analytically)

    char* p = (char*)d_ws;
    bf16* wq_t  = (bf16*)wsalloc(p, (size_t)DIM * DIM * 2);        // [N][K]
    bf16* wk_t  = (bf16*)wsalloc(p, (size_t)DIM * (NKV * HD) * 2);
    bf16* wv_t  = (bf16*)wsalloc(p, (size_t)DIM * (NKV * HD) * 2);
    bf16* wo_t  = (bf16*)wsalloc(p, (size_t)DIM * DIM * 2);
    bf16* w1_t  = (bf16*)wsalloc(p, (size_t)DIM * HID * 2);
    bf16* w2_t  = (bf16*)wsalloc(p, (size_t)HID * DIM * 2);
    bf16* w3_t  = (bf16*)wsalloc(p, (size_t)DIM * HID * 2);
    bf16* hin_h = (bf16*)wsalloc(p, (size_t)ROWS * DIM * 2);
    float* qf   = (float*)wsalloc(p, (size_t)ROWS * DIM * 4);
    float* kf   = (float*)wsalloc(p, (size_t)ROWS * (NKV * HD) * 4);
    float* vf   = (float*)wsalloc(p, (size_t)ROWS * (NKV * HD) * 4);
    bf16* Qb    = (bf16*)wsalloc(p, (size_t)ROWS * DIM * 2);
    bf16* Kb    = (bf16*)wsalloc(p, (size_t)ROWS * (NKV * HD) * 2);
    bf16* Vb    = (bf16*)wsalloc(p, (size_t)ROWS * (NKV * HD) * 2);
    bf16* atn_h = (bf16*)wsalloc(p, (size_t)ROWS * DIM * 2);
    float* hbuf = (float*)wsalloc(p, (size_t)ROWS * DIM * 4);
    bf16* g_h   = (bf16*)wsalloc(p, (size_t)ROWS * DIM * 2);
    float* g1f  = (float*)wsalloc(p, (size_t)ROWS * HID * 4);
    bf16* ffh   = (bf16*)wsalloc(p, (size_t)ROWS * HID * 2);
    (void)ws_size;

    // ---- weight prep: fp32 [K][N] -> bf16 [N][K] ----
    cvt_transpose_bf16<<<4096, 256, 0, stream>>>(wq, wq_t, DIM, DIM);
    cvt_transpose_bf16<<<2048, 256, 0, stream>>>(wk, wk_t, DIM, NKV * HD);
    cvt_transpose_bf16<<<2048, 256, 0, stream>>>(wv, wv_t, DIM, NKV * HD);
    cvt_transpose_bf16<<<4096, 256, 0, stream>>>(wo, wo_t, DIM, DIM);
    cvt_transpose_bf16<<<8192, 256, 0, stream>>>(w1, w1_t, DIM, HID);
    cvt_transpose_bf16<<<8192, 256, 0, stream>>>(w2, w2_t, HID, DIM);
    cvt_transpose_bf16<<<8192, 256, 0, stream>>>(w3, w3_t, DIM, HID);

    // ---- attention path ----
    rmsnorm_bf16<<<ROWS, 256, 0, stream>>>(x, anw, hin_h);

    gemm_bf16<<<dim3(DIM / 128, ROWS / 256), 256, 0, stream>>>(
        hin_h, wq_t, ROWS, DIM, DIM, 0, nullptr, qf, nullptr);
    gemm_bf16<<<dim3((NKV * HD) / 128, ROWS / 256), 256, 0, stream>>>(
        hin_h, wk_t, ROWS, NKV * HD, DIM, 0, nullptr, kf, nullptr);
    gemm_bf16<<<dim3((NKV * HD) / 128, ROWS / 256), 256, 0, stream>>>(
        hin_h, wv_t, ROWS, NKV * HD, DIM, 0, nullptr, vf, nullptr);

    rope_kernel<<<(BATCH * SEQ * NH * 64) / 256, 256, 0, stream>>>(qf, fcos, fsin, Qb, NH);
    rope_kernel<<<(BATCH * SEQ * NKV * 64) / 256, 256, 0, stream>>>(kf, fcos, fsin, Kb, NKV);
    vconv_kernel<<<(BATCH * SEQ * NKV * HD) / 256, 256, 0, stream>>>(vf, Vb);

    attn_kernel<<<dim3(SEQ / 128, NH, BATCH), 256, 0, stream>>>(Qb, Kb, Vb, atn_h);

    // h = x + attn @ wo
    gemm_bf16<<<dim3(DIM / 128, ROWS / 256), 256, 0, stream>>>(
        atn_h, wo_t, ROWS, DIM, DIM, 1, x, hbuf, nullptr);

    // ---- FFN path ----
    rmsnorm_bf16<<<ROWS, 256, 0, stream>>>(hbuf, fnw, g_h);

    gemm_bf16<<<dim3(HID / 128, ROWS / 256), 256, 0, stream>>>(
        g_h, w1_t, ROWS, HID, DIM, 0, nullptr, g1f, nullptr);
    // ffh = bf16( silu(g@w1) * (g@w3) )
    gemm_bf16<<<dim3(HID / 128, ROWS / 256), 256, 0, stream>>>(
        g_h, w3_t, ROWS, HID, DIM, 3, g1f, nullptr, ffh);
    // out = h + ffh @ w2
    gemm_bf16<<<dim3(DIM / 128, ROWS / 256), 256, 0, stream>>>(
        ffh, w2_t, ROWS, DIM, HID, 1, hbuf, (float*)d_out, nullptr);
    (void)in_sizes; (void)n_in; (void)out_size;
}
